// RiceTransformer_21912923144251
// MI455X (gfx1250) — matrix operand order repaired and checked by reading, not compiled, not run on hardware
//
#include <hip/hip_runtime.h>

#define S_  2048
#define D_  1024
#define H_  16
#define HD_ 64
#define FF_ 4096
#define NSEG_ 4
#define SEG_ 512
#define DEPTH_ 2

typedef unsigned short u16;
typedef __attribute__((ext_vector_type(16))) __bf16 v16bf;
typedef __attribute__((ext_vector_type(8)))  float  v8f;
typedef __attribute__((ext_vector_type(8)))  unsigned short v8u16;
typedef __attribute__((ext_vector_type(4)))  unsigned int u32x4;
typedef __attribute__((ext_vector_type(8)))  unsigned int u32x8;

// ---------- helpers ----------

__device__ inline u16 f2bf(float f) {
    union { float f; unsigned u; } c; c.f = f;
    unsigned u = c.u;
    u += 0x7FFFu + ((u >> 16) & 1u);   // round-to-nearest-even
    return (u16)(u >> 16);
}

__device__ inline v8f wmma_bf16(v16bf a, v16bf b, v8f c) {
    return __builtin_amdgcn_wmma_f32_16x16x32_bf16(
        /*neg_a=*/false, a, /*neg_b=*/false, b,
        /*c_mod=*/(short)0, c, /*reuse_a=*/false, /*reuse_b=*/false);
}

// A fragment (16x32 bf16): lane m=lane&15, g=lane>>4; K chunks [8g,8g+8) and [16+8g,..)
__device__ inline v16bf load_a_frag(const u16* base, int ld) {
    int lane = threadIdx.x & 31;
    int m = lane & 15, g = lane >> 4;
    const u16* p = base + (size_t)m * ld + 8 * g;
    union { v8u16 h[2]; v16bf v; } u;
    u.h[0] = *(const v8u16*)(p);
    u.h[1] = *(const v8u16*)(p + 16);
    return u.v;
}

// B fragment (32x16 bf16) from B^T storage [N][K]: lane n=lane&15, g=lane>>4;
// holds K chunks [8g,8g+8) and [16+8g,..) of column n (same order as A)
__device__ inline v16bf load_b_frag(const u16* base, int ld) {
    int lane = threadIdx.x & 31;
    int n = lane & 15, g = lane >> 4;
    const u16* p = base + (size_t)n * ld + 8 * g;
    union { v8u16 h[2]; v16bf v; } u;
    u.h[0] = *(const v8u16*)(p);
    u.h[1] = *(const v8u16*)(p + 16);
    return u.v;
}

__device__ inline v8f zero_v8f() {
    v8f z;
    #pragma unroll
    for (int e = 0; e < 8; e++) z[e] = 0.0f;
    return z;
}

// ---------- TDM: async 2D tile copy global -> LDS (Tensor Data Mover) ----------
// D# built per CDNA5 ISA: group0 = {count/flags, lds_addr,
// global_addr[56:0], type=2}; group1 = {mask/data_size, dims/strides}.
// tile: tile_rows rows x tile_k elems (bf16), row stride = stride_elems.
__device__ inline void tdm_load_2d(const u16* gptr, u16* lptr,
                                   int tile_k, int tile_rows, int stride_elems) {
    unsigned lds_off = (unsigned)(size_t)lptr;      // low 32 bits of flat = LDS offset
    unsigned long long ga = (unsigned long long)(size_t)gptr;
    u32x4 g0;
    g0[0] = 1u;                                      // count=1, user descriptor
    g0[1] = lds_off;                                 // lds_addr
    g0[2] = (unsigned)ga;                            // global_addr[31:0]
    g0[3] = (unsigned)((ga >> 32) & 0x01FFFFFFull)   // global_addr[56:32]
          | (2u << 30);                              // type=2 ("image")
    u32x8 g1;
    g1[0] = (1u << 16);                              // wg_mask=0, data_size=1 (2B)
    g1[1] = ((unsigned)tile_k & 0xFFFFu) << 16;      // tensor_dim0[15:0] @ bits63:48
    g1[2] = (((unsigned)tile_k >> 16) & 0xFFFFu)     // tensor_dim0[31:16]
          | (((unsigned)tile_rows & 0xFFFFu) << 16); // tensor_dim1[15:0]
    g1[3] = (((unsigned)tile_rows >> 16) & 0xFFFFu)  // tensor_dim1[31:16]
          | (((unsigned)tile_k & 0xFFFFu) << 16);    // tile_dim0
    g1[4] = ((unsigned)tile_rows & 0xFFFFu);         // tile_dim1 (tile_dim2=0)
    g1[5] = (unsigned)stride_elems;                  // tensor_dim0_stride[31:0]
    g1[6] = 0u;                                      // stride0 hi, stride1 lo
    g1[7] = 0u;
    asm volatile("tensor_load_to_lds %0, %1" :: "s"(g0), "s"(g1) : "memory");
}

__device__ inline float blk_sum(float v, float* sbuf) {
    #pragma unroll
    for (int o = 16; o > 0; o >>= 1) v += __shfl_xor(v, o, 32);
    int w = threadIdx.x >> 5, nw = blockDim.x >> 5;
    if ((threadIdx.x & 31) == 0) sbuf[w] = v;
    __syncthreads();
    float r = 0.0f;
    for (int i = 0; i < nw; i++) r += sbuf[i];
    __syncthreads();
    return r;
}

__device__ inline float blk_max(float v, float* sbuf) {
    #pragma unroll
    for (int o = 16; o > 0; o >>= 1) v = fmaxf(v, __shfl_xor(v, o, 32));
    int w = threadIdx.x >> 5, nw = blockDim.x >> 5;
    if ((threadIdx.x & 31) == 0) sbuf[w] = v;
    __syncthreads();
    float r = -3.4e38f;
    for (int i = 0; i < nw; i++) r = fmaxf(r, sbuf[i]);
    __syncthreads();
    return r;
}

// ---------- elementwise / transforms ----------

__global__ void copy_f32_kernel(const float* __restrict__ src, float* __restrict__ dst, int n) {
    int i = blockIdx.x * 256 + threadIdx.x;
    if (i < n) dst[i] = src[i];
}

// transpose + f32->bf16: src [R][C] f32 -> dst [C][R] bf16, 32x32 tiles
__global__ __launch_bounds__(256) void cvt_t_kernel(const float* __restrict__ src,
                                                    u16* __restrict__ dst, int R, int C) {
    __shared__ u16 tile[32][33];
    int tpc = C / 32;
    int tr = (blockIdx.x / tpc) * 32;
    int tc = (blockIdx.x % tpc) * 32;
    int t = threadIdx.x;
    int r = t >> 5, c = t & 31;
    #pragma unroll
    for (int p = 0; p < 4; p++)
        tile[r + p * 8][c] = f2bf(src[(size_t)(tr + r + p * 8) * C + tc + c]);
    __syncthreads();
    #pragma unroll
    for (int p = 0; p < 4; p++)
        dst[(size_t)(tc + r + p * 8) * R + tr + c] = tile[c][r + p * 8];
}

// LayerNorm over D=1024, one block (256 thr) per row; output bf16
__global__ __launch_bounds__(256) void ln_bf16_kernel(const float* __restrict__ x,
                                                      const float* __restrict__ g,
                                                      const float* __restrict__ b,
                                                      u16* __restrict__ out) {
    __shared__ float sred[8];
    int row = blockIdx.x;
    const float* xp = x + (size_t)row * D_;
    int t = threadIdx.x;
    float v[4];
    float s = 0.0f;
    #pragma unroll
    for (int i = 0; i < 4; i++) { v[i] = xp[t + i * 256]; s += v[i]; }
    s = blk_sum(s, sred);
    float mean = s * (1.0f / D_);
    float q = 0.0f;
    #pragma unroll
    for (int i = 0; i < 4; i++) { float d = v[i] - mean; q += d * d; }
    q = blk_sum(q, sred);
    float inv = rsqrtf(q * (1.0f / D_) + 1e-6f);
    #pragma unroll
    for (int i = 0; i < 4; i++) {
        int c = t + i * 256;
        out[(size_t)row * D_ + c] = f2bf((v[i] - mean) * inv * g[c] + b[c]);
    }
}

// RoPE on q,k -> bf16 [S][D]; v -> transposed bf16 vt[d][s]
__global__ void rope_split_kernel(const float* __restrict__ qkv,
                                  const float* __restrict__ cosb,
                                  const float* __restrict__ sinb,
                                  u16* __restrict__ qo, u16* __restrict__ ko,
                                  u16* __restrict__ vt) {
    int idx = blockIdx.x * 256 + threadIdx.x;
    if (idx >= S_ * D_) return;
    int s = idx / D_, d = idx % D_;
    int hd = d & (HD_ - 1);
    float c = cosb[s * HD_ + hd], sn = sinb[s * HD_ + hd];
    const float* base = qkv + (size_t)s * (3 * D_);
    float qv = base[d], kv = base[D_ + d], vv = base[2 * D_ + d];
    int dp = (hd < 32) ? (d + 32) : (d - 32);
    float qr = (hd < 32) ? -base[dp] : base[dp];
    float kr = (hd < 32) ? -base[D_ + dp] : base[D_ + dp];
    qo[idx] = f2bf(qv * c + qr * sn);
    ko[idx] = f2bf(kv * c + kr * sn);
    vt[(size_t)d * S_ + s] = f2bf(vv);
}

// ---------- WMMA GEMM: C[M][N] = A[M][K](bf16) @ B^T[N][K](bf16) + bias ----------
// mode 0: outf = val    mode 1: outf += val (residual)    mode 2: outb = bf16(gelu(val))
// Tiles staged via TDM (tensor_load_to_lds), double-buffered, waited with s_wait_tensorcnt.
#define TBM 128
#define TBN 128
#define TBK 32

__global__ __launch_bounds__(256) void gemm_bf16_kernel(const u16* __restrict__ A,
                                                        const u16* __restrict__ BT,
                                                        const float* __restrict__ bias,
                                                        float* outf, u16* outb,
                                                        int M, int N, int K, int mode) {
    __shared__ u16 sA[2][TBM][TBK];   // 16 KB
    __shared__ u16 sB[2][TBN][TBK];   // 16 KB (B^T tile: [n][k])
    int nbn = N / TBN;
    int bm = blockIdx.x / nbn, bn = blockIdx.x % nbn;
    int m0 = bm * TBM, n0 = bn * TBN;
    int tid = threadIdx.x, wave = tid >> 5, lane = tid & 31;
    int wm = wave & 3, wn = wave >> 2;       // waves 4(M) x 2(N); wave tile 32x64

    v8f acc[2][4];
    #pragma unroll
    for (int i = 0; i < 2; i++)
        #pragma unroll
        for (int j = 0; j < 4; j++) acc[i][j] = zero_v8f();

    int nk = K / TBK;
    if (wave == 0) {
        tdm_load_2d(A  + (size_t)m0 * K, &sA[0][0][0], TBK, TBM, K);
        tdm_load_2d(BT + (size_t)n0 * K, &sB[0][0][0], TBK, TBN, K);
    }
    for (int kt = 0; kt < nk; kt++) {
        int buf = kt & 1;
        if (wave == 0) {
            if (kt + 1 < nk) {
                int kn = (kt + 1) * TBK;
                tdm_load_2d(A  + (size_t)m0 * K + kn, &sA[buf ^ 1][0][0], TBK, TBM, K);
                tdm_load_2d(BT + (size_t)n0 * K + kn, &sB[buf ^ 1][0][0], TBK, TBN, K);
                asm volatile("s_wait_tensorcnt 0x2" ::: "memory");  // pair kt done
            } else {
                asm volatile("s_wait_tensorcnt 0x0" ::: "memory");
            }
        }
        __syncthreads();
        v16bf a0 = load_a_frag(&sA[buf][wm * 32][0], TBK);
        v16bf a1 = load_a_frag(&sA[buf][wm * 32 + 16][0], TBK);
        #pragma unroll
        for (int j = 0; j < 4; j++) {
            v16bf bf = load_b_frag(&sB[buf][wn * 64 + j * 16][0], TBK);
            acc[0][j] = wmma_bf16(a0, bf, acc[0][j]);
            acc[1][j] = wmma_bf16(a1, bf, acc[1][j]);
        }
        __syncthreads();
    }

    int cn = lane & 15, gg = lane >> 4;
    #pragma unroll
    for (int i = 0; i < 2; i++) {
        #pragma unroll
        for (int j = 0; j < 4; j++) {
            int tm = m0 + wm * 32 + i * 16;
            int tn = n0 + wn * 64 + j * 16;
            int col = tn + cn;
            float bv = bias ? bias[col] : 0.0f;
            #pragma unroll
            for (int r = 0; r < 8; r++) {
                int row = tm + r + 8 * gg;
                float v = acc[i][j][r] + bv;
                size_t idx = (size_t)row * N + col;
                if (mode == 0) {
                    outf[idx] = v;
                } else if (mode == 1) {
                    outf[idx] += v;
                } else {
                    float ge = 0.5f * v * (1.0f + erff(v * 0.70710678118654752f));
                    outb[idx] = f2bf(ge);
                }
            }
        }
    }
}

// ---------- attention: scores = scale * Q K^T per (head, segment) ----------
__global__ __launch_bounds__(256) void attn_scores_kernel(const u16* __restrict__ q,
                                                          const u16* __restrict__ kmat,
                                                          float* __restrict__ scores) {
    int t = blockIdx.x & 15;          // 4x4 tiles of 128 over 512x512
    int pair = blockIdx.x >> 4;       // h*NSEG + seg
    int bm = t >> 2, bn = t & 3;
    int h = pair >> 2, seg = pair & 3;
    int s0 = seg * SEG_;
    int m0 = bm * 128, n0 = bn * 128;
    int tid = threadIdx.x, wave = tid >> 5, lane = tid & 31;
    int wm = wave & 3, wn = wave >> 2;

    v8f acc[2][4];
    #pragma unroll
    for (int i = 0; i < 2; i++)
        #pragma unroll
        for (int j = 0; j < 4; j++) acc[i][j] = zero_v8f();

    const u16* qb = q    + (size_t)(s0 + m0 + wm * 32) * D_ + h * HD_;
    const u16* kb = kmat + (size_t)(s0 + n0 + wn * 64) * D_ + h * HD_;
    #pragma unroll
    for (int k0 = 0; k0 < HD_; k0 += 32) {
        v16bf a0 = load_a_frag(qb + k0, D_);
        v16bf a1 = load_a_frag(qb + (size_t)16 * D_ + k0, D_);
        #pragma unroll
        for (int j = 0; j < 4; j++) {
            v16bf bf = load_b_frag(kb + (size_t)(j * 16) * D_ + k0, D_);
            acc[0][j] = wmma_bf16(a0, bf, acc[0][j]);
            acc[1][j] = wmma_bf16(a1, bf, acc[1][j]);
        }
    }

    const float scale = 0.125f;  // 1/sqrt(64)
    int cn = lane & 15, gg = lane >> 4;
    #pragma unroll
    for (int i = 0; i < 2; i++) {
        #pragma unroll
        for (int j = 0; j < 4; j++) {
            int rseg = m0 + wm * 32 + i * 16;
            int col = n0 + wn * 64 + j * 16 + cn;
            #pragma unroll
            for (int r = 0; r < 8; r++) {
                int s = s0 + rseg + r + 8 * gg;
                scores[((size_t)h * S_ + s) * SEG_ + col] = acc[i][j][r] * scale;
            }
        }
    }
}

// softmax over 512 cols; one 128-thread block per (h, s) row; probs bf16
__global__ __launch_bounds__(128) void softmax_kernel(const float* __restrict__ scores,
                                                      u16* __restrict__ probs) {
    __shared__ float sred[4];
    size_t row = blockIdx.x;
    const float* sp = scores + row * SEG_;
    u16* pp = probs + row * SEG_;
    int t = threadIdx.x;
    float v[4];
    float mx = -3.4e38f;
    #pragma unroll
    for (int i = 0; i < 4; i++) { v[i] = sp[t + i * 128]; mx = fmaxf(mx, v[i]); }
    mx = blk_max(mx, sred);
    float sum = 0.0f;
    #pragma unroll
    for (int i = 0; i < 4; i++) { v[i] = __expf(v[i] - mx); sum += v[i]; }
    sum = blk_sum(sum, sred);
    float inv = 1.0f / sum;
    #pragma unroll
    for (int i = 0; i < 4; i++) pp[t + i * 128] = f2bf(v[i] * inv);
}

// AO = P(512x512) @ V(512x64) per (head, seg); V pre-transposed -> frags from global
__global__ __launch_bounds__(256) void attn_av_kernel(const u16* __restrict__ probs,
                                                      const u16* __restrict__ vt,
                                                      u16* __restrict__ ao) {
    int bm = blockIdx.x & 3;          // 4 row-blocks of 128 over 512
    int pair = blockIdx.x >> 2;
    int h = pair >> 2, seg = pair & 3;
    int s0 = seg * SEG_;
    int m0 = bm * 128;
    int tid = threadIdx.x, wave = tid >> 5, lane = tid & 31;
    int wrow = wave * 16;             // each wave: 16 rows x 64 cols

    v8f acc[4];
    #pragma unroll
    for (int j = 0; j < 4; j++) acc[j] = zero_v8f();

    const u16* pb = probs + ((size_t)h * S_ + s0 + m0 + wrow) * SEG_;
    const u16* vbase = vt + (size_t)h * HD_ * S_ + s0;   // row n: vt[(h*64+n)*S + s0 + k]
    #pragma unroll 4
    for (int k0 = 0; k0 < SEG_; k0 += 32) {
        v16bf a = load_a_frag(pb + k0, SEG_);
        #pragma unroll
        for (int j = 0; j < 4; j++) {
            v16bf bf = load_b_frag(vbase + (size_t)(j * 16) * S_ + k0, S_);
            acc[j] = wmma_bf16(a, bf, acc[j]);
        }
    }

    int cn = lane & 15, gg = lane >> 4;
    #pragma unroll
    for (int j = 0; j < 4; j++) {
        int col = j * 16 + cn;
        #pragma unroll
        for (int r = 0; r < 8; r++) {
            int s = s0 + m0 + wrow + r + 8 * gg;
            ao[(size_t)s * D_ + h * HD_ + col] = f2bf(acc[j][r]);
        }
    }
}

// ---------- host orchestration ----------

extern "C" void kernel_launch(void* const* d_in, const int* in_sizes, int n_in,
                              void* d_out, int out_size, void* d_ws, size_t ws_size,
                              hipStream_t stream) {
    const float* hidden = (const float*)d_in[0];
    const float* cosb   = (const float*)d_in[2];
    const float* sinb   = (const float*)d_in[3];
    const float* ln1_g  = (const float*)d_in[4];
    const float* ln1_b  = (const float*)d_in[5];
    const float* qkv_w  = (const float*)d_in[6];
    const float* qkv_b  = (const float*)d_in[7];
    const float* proj_w = (const float*)d_in[8];
    const float* proj_b = (const float*)d_in[9];
    const float* ln2_g  = (const float*)d_in[10];
    const float* ln2_b  = (const float*)d_in[11];
    const float* fc1_w  = (const float*)d_in[12];
    const float* fc1_b  = (const float*)d_in[13];
    const float* fc2_w  = (const float*)d_in[14];
    const float* fc2_b  = (const float*)d_in[15];

    float* x = (float*)d_out;          // residual stream lives in d_out
    char* ws = (char*)d_ws;
    size_t off = 0;
    auto alloc = [&](size_t bytes) {
        void* p = ws + off;
        off += (bytes + 255) & ~(size_t)255;
        return p;
    };

    // pre-transposed bf16 weights: W^T [N][K]
    u16*   wqkvT  = (u16*)alloc((size_t)DEPTH_ * 3 * D_ * D_ * 2);
    u16*   wprojT = (u16*)alloc((size_t)DEPTH_ * D_ * D_ * 2);
    u16*   wfc1T  = (u16*)alloc((size_t)DEPTH_ * FF_ * D_ * 2);
    u16*   wfc2T  = (u16*)alloc((size_t)DEPTH_ * D_ * FF_ * 2);
    u16*   hbuf   = (u16*)alloc((size_t)S_ * D_ * 2);
    float* qkvbuf = (float*)alloc((size_t)S_ * 3 * D_ * 4);
    u16*   qb     = (u16*)alloc((size_t)S_ * D_ * 2);
    u16*   kb     = (u16*)alloc((size_t)S_ * D_ * 2);
    u16*   vtb    = (u16*)alloc((size_t)S_ * D_ * 2);
    float* scores = (float*)alloc((size_t)H_ * S_ * SEG_ * 4);
    u16*   probs  = (u16*)alloc((size_t)H_ * S_ * SEG_ * 2);
    u16*   aob    = (u16*)alloc((size_t)S_ * D_ * 2);
    u16*   ffbuf  = (u16*)alloc((size_t)S_ * FF_ * 2);

    // one-time weight transpose + bf16 convert (bandwidth-trivial vs 23.3 TB/s)
    for (int i = 0; i < DEPTH_; i++) {
        cvt_t_kernel<<<(D_ / 32) * (3 * D_ / 32), 256, 0, stream>>>(
            qkv_w + (size_t)i * D_ * 3 * D_, wqkvT + (size_t)i * 3 * D_ * D_, D_, 3 * D_);
        cvt_t_kernel<<<(D_ / 32) * (D_ / 32), 256, 0, stream>>>(
            proj_w + (size_t)i * D_ * D_, wprojT + (size_t)i * D_ * D_, D_, D_);
        cvt_t_kernel<<<(D_ / 32) * (FF_ / 32), 256, 0, stream>>>(
            fc1_w + (size_t)i * D_ * FF_, wfc1T + (size_t)i * FF_ * D_, D_, FF_);
        cvt_t_kernel<<<(FF_ / 32) * (D_ / 32), 256, 0, stream>>>(
            fc2_w + (size_t)i * FF_ * D_, wfc2T + (size_t)i * D_ * FF_, FF_, D_);
    }
    // x = hidden_states
    copy_f32_kernel<<<(S_ * D_ + 255) / 256, 256, 0, stream>>>(hidden, x, S_ * D_);

    for (int i = 0; i < DEPTH_; i++) {
        ln_bf16_kernel<<<S_, 256, 0, stream>>>(x, ln1_g + i * D_, ln1_b + i * D_, hbuf);
        gemm_bf16_kernel<<<(S_ / TBM) * (3 * D_ / TBN), 256, 0, stream>>>(
            hbuf, wqkvT + (size_t)i * 3 * D_ * D_, qkv_b + (size_t)i * 3 * D_,
            qkvbuf, nullptr, S_, 3 * D_, D_, 0);
        rope_split_kernel<<<(S_ * D_ + 255) / 256, 256, 0, stream>>>(
            qkvbuf, cosb, sinb, qb, kb, vtb);
        attn_scores_kernel<<<H_ * NSEG_ * 16, 256, 0, stream>>>(qb, kb, scores);
        softmax_kernel<<<H_ * S_, 128, 0, stream>>>(scores, probs);
        attn_av_kernel<<<H_ * NSEG_ * 4, 256, 0, stream>>>(probs, vtb, aob);
        gemm_bf16_kernel<<<(S_ / TBM) * (D_ / TBN), 256, 0, stream>>>(
            aob, wprojT + (size_t)i * D_ * D_, proj_b + (size_t)i * D_,
            x, nullptr, S_, D_, D_, 1);
        ln_bf16_kernel<<<S_, 256, 0, stream>>>(x, ln2_g + i * D_, ln2_b + i * D_, hbuf);
        gemm_bf16_kernel<<<(S_ / TBM) * (FF_ / TBN), 256, 0, stream>>>(
            hbuf, wfc1T + (size_t)i * FF_ * D_, fc1_b + (size_t)i * FF_,
            nullptr, ffbuf, S_, FF_, D_, 2);
        gemm_bf16_kernel<<<(S_ / TBM) * (D_ / TBN), 256, 0, stream>>>(
            ffbuf, wfc2T + (size_t)i * D_ * FF_, fc2_b + (size_t)i * D_,
            x, nullptr, S_, D_, FF_, 1);
    }
}
